// Frame2TFNCrossAttentionUpdate_56023553409100
// MI455X (gfx1250) — matrix operand order repaired and checked by reading, not compiled, not run on hardware
//
#include <hip/hip_runtime.h>
#include <hip/hip_fp16.h>

typedef __attribute__((ext_vector_type(16))) _Float16 v16h;
typedef __attribute__((ext_vector_type(8)))  float    v8f;
typedef __attribute__((ext_vector_type(4)))  int      v4i;

static constexpr int Nn = 10000;   // nodes
static constexpr int Ee = 160000;  // edges

// ---------------- float <-> ordered-uint keys (for atomic float max) ----------
__device__ __forceinline__ unsigned fkey(float f) {
  unsigned b = __float_as_uint(f);
  return (b & 0x80000000u) ? ~b : (b | 0x80000000u);
}
__device__ __forceinline__ float funkey(unsigned k) {
  unsigned b = (k & 0x80000000u) ? (k & 0x7fffffffu) : ~k;
  return __uint_as_float(b);
}

// ---------------- kernel 1: convert weights into f16 WMMA B-fragments ---------
// B 32x16 f16 layout: lanes 0-15 -> N=lane, K=0..7,16..23; lanes 16-31 -> N=lane-16, K=8..15,24..31
// Stored as [tile][lane][16 halfs] contiguous; tiles packed: Wk1(2) Wv1(2) Wk2(40) Wv2(56)
__global__ void conv_wfrags(const float* __restrict__ Wk1, const float* __restrict__ Wv1,
                            const float* __restrict__ Wk2, const float* __restrict__ Wv2,
                            _Float16* __restrict__ wfragG) {
  int tid  = blockIdx.x * blockDim.x + threadIdx.x;
  int t    = tid >> 5;
  int lane = tid & 31;
  if (t >= 100) return;
  const float* W; int NC, lt;
  if (t < 2)       { W = Wk1; NC = 32;  lt = t;      }
  else if (t < 4)  { W = Wv1; NC = 32;  lt = t - 2;  }
  else if (t < 44) { W = Wk2; NC = 640; lt = t - 4;  }
  else             { W = Wv2; NC = 896; lt = t - 44; }
  int ncol = lt * 16 + (lane & 15);
  int kb   = (lane < 16) ? 0 : 8;
  #pragma unroll
  for (int j = 0; j < 16; ++j)
    wfragG[((size_t)(t * 32 + lane)) * 16 + j] = (_Float16)W[(size_t)(kb + j + (j >= 8 ? 8 : 0)) * NC + ncol];
}

// ---------------- kernel 2: node precompute (f0,f1,rot,gs,gv,q) ---------------
__global__ __launch_bounds__(256) void node_pre(
    const float* __restrict__ ff,   const float* __restrict__ rot,
    const float* __restrict__ trans,const float* __restrict__ tfn,
    const float* __restrict__ Wf0,  const float* __restrict__ bf0,
    const float* __restrict__ Wf1,  const float* __restrict__ bf1,
    const float* __restrict__ Wgs,  const float* __restrict__ Wgv,
    const float* __restrict__ Wqs,  const float* __restrict__ Wqv,
    float* __restrict__ gs, float* __restrict__ gv, float* __restrict__ q) {
  __shared__ float ffL[128], f0L[64], f1L[96], f1R[96];
  const int n = blockIdx.x;
  const int t = threadIdx.x;
  if (t < 128) ffL[t] = ff[(size_t)n * 128 + t];
  __syncthreads();
  if (t < 64) {                       // f0 = ff @ W_f0 + b
    float acc = bf0[t];
    for (int k = 0; k < 128; ++k) acc += ffL[k] * Wf0[k * 64 + t];
    f0L[t] = acc;
  } else if (t < 160) {               // f1 = ff @ W_f1 + b
    int c = t - 64;
    float acc = bf1[c];
    for (int k = 0; k < 128; ++k) acc += ffL[k] * Wf1[k * 96 + c];
    f1L[c] = acc;
  }
  __syncthreads();
  if (t < 96) {                       // rotate + translate
    int u = t / 3, i = t % 3;
    const float* R = rot + (size_t)n * 9;
    f1R[t] = R[i * 3 + 0] * f1L[u * 3 + 0] + R[i * 3 + 1] * f1L[u * 3 + 1]
           + R[i * 3 + 2] * f1L[u * 3 + 2] + trans[n * 3 + i];
  }
  __syncthreads();
  if (t < 16) {                       // gs = f0 @ Wg_s / sqrt(64)
    float acc = 0.f;
    for (int k = 0; k < 64; ++k) acc += f0L[k] * Wgs[k * 16 + t];
    gs[(size_t)n * 16 + t] = acc * 0.125f;
  } else if (t < 40) {                // gv = f1R @ Wg_v / sqrt(32)
    int idx = t - 16, w = idx / 3, i = idx % 3;
    float acc = 0.f;
    for (int u = 0; u < 32; ++u) acc += f1R[u * 3 + i] * Wgv[u * 8 + w];
    gv[(size_t)n * 24 + w * 3 + i] = acc * 0.1767766953f;
  } else if (t < 56) {                // q_s = tfn_s @ Wq_s / 4
    int c = t - 40;
    float acc = 0.f;
    const float* ts = tfn + (size_t)n * 40;
    for (int u = 0; u < 16; ++u) acc += ts[u] * Wqs[u * 16 + c];
    q[(size_t)n * 40 + c] = acc * 0.25f;
  } else if (t < 80) {                // q_v = tfn_v @ Wq_v / sqrt(8)
    int idx = t - 56, w = idx / 3, i = idx % 3;
    float acc = 0.f;
    const float* tv = tfn + (size_t)n * 40 + 16;
    for (int u = 0; u < 8; ++u) acc += tv[u * 3 + i] * Wqv[u * 8 + w];
    q[(size_t)n * 40 + 16 + w * 3 + i] = acc * 0.3535533906f;
  }
}

// ---------------- kernel 3: init softmax/aggregation buffers ------------------
__global__ void init_buffers(unsigned* __restrict__ mrow, float* __restrict__ den,
                             float* __restrict__ upds, float* __restrict__ updv,
                             float* __restrict__ stats) {
  int i = blockIdx.x * blockDim.x + threadIdx.x;
  if (i < Nn * 4)  { mrow[i] = 0x00800000u; den[i] = 0.f; }  // key(-FLT_MAX)
  if (i < Nn * 16) upds[i] = 0.f;
  if (i < Nn * 48) updv[i] = 0.f;
  if (i < 40)      stats[i] = 0.f;
}

// ---------------- kernel 4: fused edge kernel (WMMA) --------------------------
// 4 waves/block, 16 edges/wave. All f16 weight fragments are async-staged into
// LDS once per block (102400 B) and shared by the 4 waves; WMMA B operands then
// come from ds_load_b128. TP folded into the WMMA tile loop; attention logits
// via LDS float atomics; v written to workspace.
__global__ __launch_bounds__(128) void edge_kernel(
    const float* __restrict__ ef,  const float* __restrict__ esh,
    const int*   __restrict__ eidx,
    const float* __restrict__ gs,  const float* __restrict__ gv,
    const float* __restrict__ q,
    const _Float16* __restrict__ wfragG,
    const float* __restrict__ bk1, const float* __restrict__ bk2,
    const float* __restrict__ bv1, const float* __restrict__ bv2,
    float* __restrict__ attnB, float* __restrict__ vbuf,
    unsigned int* __restrict__ mrow) {
  __shared__ __align__(32) _Float16 wfragL[100 * 512];   // 102400 B
  __shared__ float cA[4][16][16], cB[4][16][8], cC[4][16][16];
  __shared__ float cD[4][16][24], cE[4][16][24];
  __shared__ float shvL[4][16][3], qS[4][16][16], qV[4][16][24], attnL[4][16][4];
  __shared__ int srcL[4][16];
  __shared__ _Float16 hbuf[4][16][32];

  const int tid  = (int)threadIdx.x;
  const int wv   = tid >> 5;                // wave id within block
  const int lane = tid & 31;
  const int e0   = (int)blockIdx.x * 64 + wv * 16;
  const int c    = lane & 15;               // D-tile column / frag row
  const int eoff = (lane < 16) ? 0 : 8;     // C layout: vgpr r -> row r (+8)

  // ---- stage all weight fragments global -> LDS (async if available) ---------
#if __has_builtin(__builtin_amdgcn_global_load_async_to_lds_b128)
  for (int idx = tid; idx < 3200; idx += 128) {
    __builtin_amdgcn_global_load_async_to_lds_b128(
        (v4i*)(wfragG + (size_t)idx * 16),
        (v4i*)(wfragL + (size_t)idx * 16),
        0, 0);
  }
#else
  for (int idx = tid; idx < 3200; idx += 128) {
    v16h tmp = *(const v16h*)(wfragG + (size_t)idx * 16);
    *(v16h*)(wfragL + (size_t)idx * 16) = tmp;
  }
#endif

  // hint the vector path to start pulling the edge features while we stage
  __builtin_prefetch(ef + (size_t)(e0 + c) * 32, 0, 3);

  // ---- per-edge TP coefficients + q gather (by src), dst-features gather -----
  if (lane < 16) {
    const int e  = e0 + lane;
    const int dn = eidx[e];
    const int sn = eidx[Ee + e];
    srcL[wv][lane] = sn;
    const float shs = esh[(size_t)e * 4 + 0];
    const float sv0 = esh[(size_t)e * 4 + 1];
    const float sv1 = esh[(size_t)e * 4 + 2];
    const float sv2 = esh[(size_t)e * 4 + 3];
    shvL[wv][lane][0] = sv0; shvL[wv][lane][1] = sv1; shvL[wv][lane][2] = sv2;
    const float* g0 = gs + (size_t)dn * 16;
    #pragma unroll
    for (int u = 0; u < 16; ++u) {
      float f = g0[u];
      cA[wv][lane][u] = f * shs * 0.25f;    // 0e*0e path, /sqrt(16)
      cC[wv][lane][u] = f * 0.25f;          // 0e*1o path, /sqrt(16)
    }
    const float* g1 = gv + (size_t)dn * 24;
    const float i8 = 0.3535533906f;         // 1/sqrt(8)
    #pragma unroll
    for (int u = 0; u < 8; ++u) {
      float x = g1[u * 3 + 0], y = g1[u * 3 + 1], z = g1[u * 3 + 2];
      cB[wv][lane][u] = (x * sv0 + y * sv1 + z * sv2) * i8;  // 1o.1o dot
      cD[wv][lane][u * 3 + 0] = x * shs * i8;                // 1o*0e
      cD[wv][lane][u * 3 + 1] = y * shs * i8;
      cD[wv][lane][u * 3 + 2] = z * shs * i8;
      cE[wv][lane][u * 3 + 0] = (y * sv2 - z * sv1) * i8;    // 1o x 1o (cross)
      cE[wv][lane][u * 3 + 1] = (z * sv0 - x * sv2) * i8;
      cE[wv][lane][u * 3 + 2] = (x * sv1 - y * sv0) * i8;
    }
    const float* qq = q + (size_t)sn * 40;
    #pragma unroll
    for (int j = 0; j < 16; ++j) qS[wv][lane][j] = qq[j];
    #pragma unroll
    for (int j = 0; j < 24; ++j) qV[wv][lane][j] = qq[16 + j];
    #pragma unroll
    for (int h = 0; h < 4; ++h) attnL[wv][lane][h] = 0.f;
  }

  // ---- build edge-feature A fragment (16x32 f16, A layout) directly ----------
  v16h a_ef;
  {
    const float* rp = ef + (size_t)(e0 + c) * 32;
    const int kb = (lane < 16) ? 0 : 8;
    #pragma unroll
    for (int r = 0; r < 4; ++r) {
      float2 f = *(const float2*)(rp + kb + 2 * r);
      a_ef[2 * r]     = (_Float16)f.x;
      a_ef[2 * r + 1] = (_Float16)f.y;
      float2 g = *(const float2*)(rp + 16 + kb + 2 * r);
      a_ef[8 + 2 * r] = (_Float16)g.x;
      a_ef[9 + 2 * r] = (_Float16)g.y;
    }
  }

  // weights must be resident in LDS before the first WMMA
  asm volatile("s_wait_asynccnt 0x0" ::: "memory");
  __syncthreads();

  const _Float16* w1kL = wfragL;             // tiles 0..1
  const _Float16* w1vL = wfragL + 2 * 512;   // tiles 2..3
  const _Float16* w2kL = wfragL + 4 * 512;   // tiles 4..43
  const _Float16* w2vL = wfragL + 44 * 512;  // tiles 44..99

  // ---- layer-1 GEMMs: h = relu(ef@W1 + b); relayout D->A frag via LDS --------
  v16h ah_k, ah_v;
  #pragma unroll
  for (int which = 0; which < 2; ++which) {
    const _Float16* wf = which ? w1vL : w1kL;
    const float*    bb = which ? bv1  : bk1;
    #pragma unroll
    for (int tt = 0; tt < 2; ++tt) {
      v16h b = *(const v16h*)(wf + (size_t)(tt * 32 + lane) * 16);
      v8f d = {};
      d = __builtin_amdgcn_wmma_f32_16x16x32_f16(false, a_ef, false, b, (short)0, d, false, false);
      float bias = bb[tt * 16 + c];
      #pragma unroll
      for (int r = 0; r < 8; ++r)
        hbuf[wv][r + eoff][tt * 16 + c] = (_Float16)fmaxf(d[r] + bias, 0.0f);
    }
    __syncthreads();
    v16h ah;
    {
      const _Float16* hr = &hbuf[wv][c][0];
      const int kb = (lane < 16) ? 0 : 8;
      #pragma unroll
      for (int r = 0; r < 4; ++r) {
        ah[2 * r]     = hr[kb + 2 * r];
        ah[2 * r + 1] = hr[kb + 2 * r + 1];
        ah[8 + 2 * r] = hr[16 + kb + 2 * r];
        ah[9 + 2 * r] = hr[16 + kb + 2 * r + 1];
      }
    }
    if (which) ah_v = ah; else ah_k = ah;
    __syncthreads();
  }

  const float inv_s2 = 0.7071067811865476f;
  const float inv_s3 = 0.5773502691896258f;

  float accS[8], accC[8], accV[8][3];
  #pragma unroll
  for (int r = 0; r < 8; ++r) {
    accS[r] = 0.f; accC[r] = 0.f;
    accV[r][0] = accV[r][1] = accV[r][2] = 0.f;
  }

  // ================= K tensor product: 40 tiles of kw = h_k @ Wk2 =============
  #define K_TILE(T)                                                                 \
    v16h b = *(const v16h*)(w2kL + (size_t)((T) * 32 + lane) * 16);                 \
    v8f d = {};                                                                     \
    d = __builtin_amdgcn_wmma_f32_16x16x32_f16(false, ah_k, false, b, (short)0, d,  \
                                               false, false);                       \
    float bias = bk2[(T) * 16 + c];
  for (int t = 0; t < 16; ++t) {            // path A: u = t
    K_TILE(t)
    #pragma unroll
    for (int r = 0; r < 8; ++r) accS[r] += cA[wv][r + eoff][t] * (d[r] + bias);
  }
  for (int t = 16; t < 24; ++t) {           // path B: u = t-16
    K_TILE(t)
    #pragma unroll
    for (int r = 0; r < 8; ++r) accS[r] += cB[wv][r + eoff][t - 16] * (d[r] + bias);
  }
  for (int t = 24; t < 32; ++t) {           // path C: col packs (u,w), o1=8
    K_TILE(t)
    int u = ((t - 24) << 1) | (c >> 3);
    #pragma unroll
    for (int r = 0; r < 8; ++r) accC[r] += cC[wv][r + eoff][u] * (d[r] + bias);
  }
  for (int t = 32; t < 36; ++t) {           // path D
    K_TILE(t)
    int u = ((t - 32) << 1) | (c >> 3);
    #pragma unroll
    for (int r = 0; r < 8; ++r) {
      float val = d[r] + bias;
      #pragma unroll
      for (int i = 0; i < 3; ++i) accV[r][i] += cD[wv][r + eoff][u * 3 + i] * val;
    }
  }
  for (int t = 36; t < 40; ++t) {           // path E
    K_TILE(t)
    int u = ((t - 36) << 1) | (c >> 3);
    #pragma unroll
    for (int r = 0; r < 8; ++r) {
      float val = d[r] + bias;
      #pragma unroll
      for (int i = 0; i < 3; ++i) accV[r][i] += cE[wv][r + eoff][u * 3 + i] * val;
    }
  }
  #undef K_TILE

  // ---- attention logits: attn[e][h] += q.k  (linear in k partials) -----------
  {
    const int wvi = c & 7;
    #pragma unroll
    for (int r = 0; r < 8; ++r) {
      const int e = r + eoff;
      float ks = accS[r] * inv_s2;
      atomicAdd(&attnL[wv][e][c >> 2], qS[wv][e][c] * ks);
      float p2 = 0.f;
      #pragma unroll
      for (int i = 0; i < 3; ++i) {
        float kv = (accC[r] * shvL[wv][e][i] + accV[r][i]) * inv_s3;
        p2 += qV[wv][e][wvi * 3 + i] * kv;
      }
      atomicAdd(&attnL[wv][e][wvi >> 1], p2);
    }
  }
  __syncthreads();
  if (lane < 16) {
    const int s = srcL[wv][lane];
    #pragma unroll
    for (int h = 0; h < 4; ++h) {
      float a = attnL[wv][lane][h];
      attnB[(size_t)(e0 + lane) * 4 + h] = a;
      atomicMax(mrow + (size_t)s * 4 + h, fkey(a));   // segment max over src
    }
  }

  // ================= V tensor product: 56 tiles of vw = h_v @ Wv2 =============
  #pragma unroll
  for (int r = 0; r < 8; ++r) {
    accS[r] = 0.f; accC[r] = 0.f;
    accV[r][0] = accV[r][1] = accV[r][2] = 0.f;
  }
  #define V_TILE(T)                                                                 \
    v16h b = *(const v16h*)(w2vL + (size_t)((T) * 32 + lane) * 16);                 \
    v8f d = {};                                                                     \
    d = __builtin_amdgcn_wmma_f32_16x16x32_f16(false, ah_v, false, b, (short)0, d,  \
                                               false, false);                       \
    float bias = bv2[(T) * 16 + c];
  for (int t = 0; t < 16; ++t) {            // A': u = t
    V_TILE(t)
    #pragma unroll
    for (int r = 0; r < 8; ++r) accS[r] += cA[wv][r + eoff][t] * (d[r] + bias);
  }
  for (int t = 16; t < 24; ++t) {           // B'
    V_TILE(t)
    #pragma unroll
    for (int r = 0; r < 8; ++r) accS[r] += cB[wv][r + eoff][t - 16] * (d[r] + bias);
  }
  for (int t = 24; t < 40; ++t) {           // C': o1=16 -> one u per tile
    V_TILE(t)
    #pragma unroll
    for (int r = 0; r < 8; ++r) accC[r] += cC[wv][r + eoff][t - 24] * (d[r] + bias);
  }
  for (int t = 40; t < 48; ++t) {           // D'
    V_TILE(t)
    #pragma unroll
    for (int r = 0; r < 8; ++r) {
      float val = d[r] + bias;
      #pragma unroll
      for (int i = 0; i < 3; ++i) accV[r][i] += cD[wv][r + eoff][(t - 40) * 3 + i] * val;
    }
  }
  for (int t = 48; t < 56; ++t) {           // E'
    V_TILE(t)
    #pragma unroll
    for (int r = 0; r < 8; ++r) {
      float val = d[r] + bias;
      #pragma unroll
      for (int i = 0; i < 3; ++i) accV[r][i] += cE[wv][r + eoff][(t - 48) * 3 + i] * val;
    }
  }
  #undef V_TILE

  // ---- store v per edge: [16 scalar | 16x3 vector] ---------------------------
  #pragma unroll
  for (int r = 0; r < 8; ++r) {
    const int e = r + eoff;
    float* vp = vbuf + (size_t)(e0 + e) * 64;
    vp[c] = accS[r] * inv_s2;
    #pragma unroll
    for (int i = 0; i < 3; ++i)
      vp[16 + c * 3 + i] = (accC[r] * shvL[wv][e][i] + accV[r][i]) * inv_s3;
  }
}

// ---------------- kernel 5: softmax numerators + segment aggregation ----------
__global__ void soft_agg(const float* __restrict__ attnB, const unsigned* __restrict__ mrow,
                         const float* __restrict__ vbuf, const int* __restrict__ eidx,
                         float* __restrict__ den, float* __restrict__ upds,
                         float* __restrict__ updv) {
  int e = blockIdx.x * blockDim.x + threadIdx.x;
  if (e >= Ee) return;
  const int s = eidx[Ee + e];
  float ex[4];
  #pragma unroll
  for (int h = 0; h < 4; ++h) {
    float m = funkey(mrow[(size_t)s * 4 + h]);
    ex[h] = __expf(attnB[(size_t)e * 4 + h] - m);
    atomicAdd(&den[(size_t)s * 4 + h], ex[h]);
  }
  const float* v = vbuf + (size_t)e * 64;
  #pragma unroll
  for (int w = 0; w < 16; ++w)
    atomicAdd(&upds[(size_t)s * 16 + w], v[w] * ex[w >> 2]);
  #pragma unroll
  for (int w = 0; w < 16; ++w) {
    float a = ex[w >> 2];
    #pragma unroll
    for (int i = 0; i < 3; ++i)
      atomicAdd(&updv[(size_t)s * 48 + w * 3 + i], v[16 + w * 3 + i] * a);
  }
}

// ---------------- kernel 6: per-node output projections + BN statistics -------
__global__ void node_fin(const float* __restrict__ den, const float* __restrict__ upds,
                         const float* __restrict__ updv, const float* __restrict__ tfn,
                         const float* __restrict__ Wos, const float* __restrict__ Wov,
                         const float* __restrict__ Wss, const float* __restrict__ Wsv,
                         float* __restrict__ outs, float* __restrict__ outv,
                         float* __restrict__ stats) {
  int n = blockIdx.x * blockDim.x + threadIdx.x;
  if (n >= Nn) return;
  float dn[4];
  #pragma unroll
  for (int h = 0; h < 4; ++h) dn[h] = den[(size_t)n * 4 + h] + 1e-9f;
  float as[16], av[16][3];
  #pragma unroll
  for (int w = 0; w < 16; ++w) {
    float inv = 1.f / dn[w >> 2];
    as[w] = upds[(size_t)n * 16 + w] * inv;
    #pragma unroll
    for (int i = 0; i < 3; ++i) av[w][i] = updv[(size_t)n * 48 + w * 3 + i] * inv;
  }
  const float* ts = tfn + (size_t)n * 40;
  for (int cc = 0; cc < 16; ++cc) {
    float o = 0.f, o2 = 0.f;
    for (int w = 0; w < 16; ++w) o  += as[w] * Wos[w * 16 + cc];
    for (int u = 0; u < 16; ++u) o2 += ts[u] * Wss[u * 16 + cc];
    o = o * 0.25f + o2 * 0.25f;                       // /sqrt(16) each
    outs[(size_t)n * 16 + cc] = o;
    atomicAdd(&stats[cc], o);
    atomicAdd(&stats[16 + cc], o * o);
  }
  for (int cc = 0; cc < 8; ++cc) {
    float nv2 = 0.f;
    #pragma unroll
    for (int i = 0; i < 3; ++i) {
      float o = 0.f, o2 = 0.f;
      for (int w = 0; w < 16; ++w) o  += av[w][i] * Wov[w * 8 + cc];
      for (int u = 0; u < 8; ++u)  o2 += ts[16 + u * 3 + i] * Wsv[u * 8 + cc];
      o = o * 0.25f + o2 * 0.3535533906f;             // /sqrt(16), /sqrt(8)
      outv[(size_t)n * 24 + cc * 3 + i] = o;
      nv2 += o * o;
    }
    atomicAdd(&stats[32 + cc], nv2);
  }
}

// ---------------- kernel 7: batch-norm finalize + concat ----------------------
__global__ void bn_out(const float* __restrict__ outs, const float* __restrict__ outv,
                       const float* __restrict__ stats, const float* __restrict__ bnws,
                       const float* __restrict__ bnbs, const float* __restrict__ bnwv,
                       float* __restrict__ out) {
  int n = blockIdx.x * blockDim.x + threadIdx.x;
  if (n >= Nn) return;
  const float invN = 1.0f / (float)Nn;
  #pragma unroll
  for (int cc = 0; cc < 16; ++cc) {
    float mu  = stats[cc] * invN;
    float var = stats[16 + cc] * invN - mu * mu;
    float x   = (outs[(size_t)n * 16 + cc] - mu) * rsqrtf(var + 1e-5f);
    out[(size_t)n * 40 + cc] = x * bnws[cc] + bnbs[cc];
  }
  #pragma unroll
  for (int u = 0; u < 8; ++u) {
    float sc = bnwv[u] * rsqrtf(stats[32 + u] * invN + 1e-5f);
    #pragma unroll
    for (int i = 0; i < 3; ++i)
      out[(size_t)n * 40 + 16 + u * 3 + i] = outv[(size_t)n * 24 + u * 3 + i] * sc;
  }
}

// ---------------- host launcher ----------------------------------------------
extern "C" void kernel_launch(void* const* d_in, const int* in_sizes, int n_in,
                              void* d_out, int out_size, void* d_ws, size_t ws_size,
                              hipStream_t stream) {
  (void)in_sizes; (void)n_in; (void)out_size; (void)ws_size;
  const float* ff    = (const float*)d_in[0];
  const float* rot   = (const float*)d_in[1];
  const float* trans = (const float*)d_in[2];
  const float* tfn   = (const float*)d_in[3];
  const float* ef    = (const float*)d_in[4];
  const float* esh   = (const float*)d_in[5];
  const int*   eidx  = (const int*)  d_in[6];
  const float* Wf0   = (const float*)d_in[7];
  const float* bf0   = (const float*)d_in[8];
  const float* Wf1   = (const float*)d_in[9];
  const float* bf1   = (const float*)d_in[10];
  const float* Wgs   = (const float*)d_in[11];
  const float* Wgv   = (const float*)d_in[12];
  const float* Wqs   = (const float*)d_in[13];
  const float* Wqv   = (const float*)d_in[14];
  const float* Wk1   = (const float*)d_in[15];
  const float* bk1   = (const float*)d_in[16];
  const float* Wk2   = (const float*)d_in[17];
  const float* bk2   = (const float*)d_in[18];
  const float* Wv1   = (const float*)d_in[19];
  const float* bv1   = (const float*)d_in[20];
  const float* Wv2   = (const float*)d_in[21];
  const float* bv2   = (const float*)d_in[22];
  const float* Wos   = (const float*)d_in[23];
  const float* Wov   = (const float*)d_in[24];
  const float* Wss   = (const float*)d_in[25];
  const float* Wsv   = (const float*)d_in[26];
  const float* bnws  = (const float*)d_in[27];
  const float* bnbs  = (const float*)d_in[28];
  const float* bnwv  = (const float*)d_in[29];
  float* out = (float*)d_out;

  // workspace layout
  _Float16* wfragG = (_Float16*)d_ws;               // 100 tiles * 512 halfs
  float* fb    = (float*)(wfragG + 100 * 512);
  float* gsB   = fb;
  float* gvB   = gsB  + (size_t)Nn * 16;
  float* qB    = gvB  + (size_t)Nn * 24;
  float* attnB = qB   + (size_t)Nn * 40;
  float* vbuf  = attnB+ (size_t)Ee * 4;
  unsigned* mrow = (unsigned*)(vbuf + (size_t)Ee * 64);
  float* den   = (float*)(mrow + (size_t)Nn * 4);
  float* upds  = den  + (size_t)Nn * 4;
  float* updv  = upds + (size_t)Nn * 16;
  float* outs  = updv + (size_t)Nn * 48;
  float* outv  = outs + (size_t)Nn * 24;
  float* stats = outv + (size_t)Nn * 24;
  // note: outs region is Nn*16 floats; gap above keeps alignment generous
  outs = updv + (size_t)Nn * 48;
  outv = outs + (size_t)Nn * 16;
  stats = outv + (size_t)Nn * 24;

  conv_wfrags<<<(100 * 32 + 255) / 256, 256, 0, stream>>>(Wk1, Wv1, Wk2, Wv2, wfragG);
  node_pre<<<Nn, 256, 0, stream>>>(ff, rot, trans, tfn, Wf0, bf0, Wf1, bf1,
                                   Wgs, Wgv, Wqs, Wqv, gsB, gvB, qB);
  init_buffers<<<(Nn * 48 + 255) / 256, 256, 0, stream>>>(mrow, den, upds, updv, stats);
  edge_kernel<<<Ee / 64, 128, 0, stream>>>(ef, esh, eidx, gsB, gvB, qB, wfragG,
                                           bk1, bk2, bv1, bv2, attnB, vbuf, mrow);
  soft_agg<<<(Ee + 255) / 256, 256, 0, stream>>>(attnB, mrow, vbuf, eidx, den, upds, updv);
  node_fin<<<(Nn + 255) / 256, 256, 0, stream>>>(den, upds, updv, tfn, Wos, Wov, Wss, Wsv,
                                                 outs, outv, stats);
  bn_out<<<(Nn + 255) / 256, 256, 0, stream>>>(outs, outv, stats, bnws, bnbs, bnwv, out);
}
